// Llama32Attention_14757507629132
// MI455X (gfx1250) — compile-verified, matrix operand order checked by reading, not run on hardware
//
#include <hip/hip_runtime.h>
#include <hip/hip_bf16.h>

// ---------------------------------------------------------------------------
// Llama-3.2-style attention layer on gfx1250 (CDNA5), bf16 WMMA everywhere.
// ---------------------------------------------------------------------------

#define T_SEQ     2048
#define HIDDEN_W  2048
#define NUM_H     32
#define NUM_KV    8
#define HEAD_D    64
#define QKV_N     ((NUM_H + 2 * NUM_KV) * HEAD_D)   // 3072
#define Q_SZ      (NUM_H * HEAD_D)                  // 2048
#define KV_SZ     (NUM_KV * HEAD_D)                 // 512
#define ROPE_TH   500000.0f
#define SM_SCALE  0.125f                            // 64^-0.5

typedef __bf16 v16bf __attribute__((ext_vector_type(16)));
typedef __bf16 v8bf  __attribute__((ext_vector_type(8)));
typedef float  v8f   __attribute__((ext_vector_type(8)));

// Per-lane A/B fragment for V_WMMA_F32_16X16X32_BF16:
// lane holds two contiguous 8-element K runs at +0 and +16 elements.
__device__ __forceinline__ v16bf load_ab_frag(const __bf16* p) {
    const v8bf* pv = (const v8bf*)__builtin_assume_aligned(p, 16);
    v8bf lo = pv[0];          // K = base .. base+7
    v8bf hi = pv[2];          // K = base+16 .. base+23
    return __builtin_shufflevector(lo, hi, 0, 1, 2, 3, 4, 5, 6, 7,
                                           8, 9, 10, 11, 12, 13, 14, 15);
}

__device__ __forceinline__ v8f wmma_bf16(v16bf a, v16bf b, v8f c) {
    return __builtin_amdgcn_wmma_f32_16x16x32_bf16(
        /*neg_a=*/false, a, /*neg_b=*/false, b,
        /*c_mod=*/(short)0, c, /*reuse_a=*/false, /*reuse_b=*/false);
}

__device__ __forceinline__ float rmax16(float v) {
    v = fmaxf(v, __shfl_xor(v, 1, 32));
    v = fmaxf(v, __shfl_xor(v, 2, 32));
    v = fmaxf(v, __shfl_xor(v, 4, 32));
    v = fmaxf(v, __shfl_xor(v, 8, 32));
    return v;
}
__device__ __forceinline__ float rsum16(float v) {
    v += __shfl_xor(v, 1, 32);
    v += __shfl_xor(v, 2, 32);
    v += __shfl_xor(v, 4, 32);
    v += __shfl_xor(v, 8, 32);
    return v;
}

// ---------------------------------------------------------------------------
// fp32 -> bf16 cast
// ---------------------------------------------------------------------------
__global__ void cast_f32_to_bf16(const float* __restrict__ src,
                                 __bf16* __restrict__ dst, int n) {
    int i = blockIdx.x * blockDim.x + threadIdx.x;
    if (i < n) dst[i] = (__bf16)src[i];
}

// ---------------------------------------------------------------------------
// GEMM: C[M,N] = A[M,K] (bf16, row-major) * B[N,K]^T (bf16, row-major).
// One wave32 computes a 32x32 output tile with 4 bf16 WMMAs per K=32 step.
// Writes fp32 (Cf) or bf16 (Cb).
// ---------------------------------------------------------------------------
__global__ void __launch_bounds__(256)
gemm_bf16_nt(const __bf16* __restrict__ A, const __bf16* __restrict__ B,
             float* __restrict__ Cf, __bf16* __restrict__ Cb,
             int M, int N, int K) {
    const int lane = threadIdx.x & 31;
    const int lr = lane & 15;       // M (or N) within 16-tile
    const int g  = lane >> 4;       // K-run group
    const int wid = (blockIdx.x * blockDim.x + threadIdx.x) >> 5;
    const int tilesN = N >> 5;
    const int tm = wid / tilesN;
    const int tn = wid % tilesN;
    if (tm * 32 >= M) return;

    v8f acc[2][2] = {};
    const __bf16* a0p = A + (size_t)(tm * 32 + lr) * K + g * 8;
    const __bf16* a1p = a0p + (size_t)16 * K;
    const __bf16* b0p = B + (size_t)(tn * 32 + lr) * K + g * 8;
    const __bf16* b1p = b0p + (size_t)16 * K;

    for (int k0 = 0; k0 < K; k0 += 32) {
        v16bf a0 = load_ab_frag(a0p + k0);
        v16bf a1 = load_ab_frag(a1p + k0);
        v16bf b0 = load_ab_frag(b0p + k0);
        v16bf b1 = load_ab_frag(b1p + k0);
        acc[0][0] = wmma_bf16(a0, b0, acc[0][0]);
        acc[0][1] = wmma_bf16(a0, b1, acc[0][1]);
        acc[1][0] = wmma_bf16(a1, b0, acc[1][0]);
        acc[1][1] = wmma_bf16(a1, b1, acc[1][1]);
    }

#pragma unroll
    for (int i = 0; i < 2; ++i)
#pragma unroll
        for (int j = 0; j < 2; ++j)
#pragma unroll
            for (int r = 0; r < 8; ++r) {
                int row = tm * 32 + i * 16 + r + 8 * g;   // C layout: M = r + 8g
                int col = tn * 32 + j * 16 + lr;          //           N = lane%16
                if (Cf) Cf[(size_t)row * N + col] = acc[i][j][r];
                else    Cb[(size_t)row * N + col] = (__bf16)acc[i][j][r];
            }
}

// ---------------------------------------------------------------------------
// Neox RoPE on Q and K slices of the packed qkv buffer.
// One thread per (t, head, pair-index).  heads 0..31 = Q, 32..39 = K.
// ---------------------------------------------------------------------------
__global__ void rope_kernel(const __bf16* __restrict__ qkv,
                            const int* __restrict__ positions,
                            __bf16* __restrict__ q_rope,
                            __bf16* __restrict__ k_rope) {
    int idx = blockIdx.x * blockDim.x + threadIdx.x;
    const int total = T_SEQ * (NUM_H + NUM_KV) * (HEAD_D / 2);
    if (idx >= total) return;
    int i    = idx & 31;                       // rotary pair index 0..31
    int head = (idx >> 5) % (NUM_H + NUM_KV);
    int t    = idx / ((NUM_H + NUM_KV) * 32);

    float p = (float)positions[t];
    float inv_freq = __powf(ROPE_TH, -((2.0f * (float)i) / (float)HEAD_D));
    float s, c;
    __sincosf(p * inv_freq, &s, &c);

    if (head < NUM_H) {
        const __bf16* src = qkv + (size_t)t * QKV_N + head * HEAD_D;
        float x1 = (float)src[i], x2 = (float)src[i + 32];
        __bf16* dst = q_rope + (size_t)t * Q_SZ + head * HEAD_D;
        dst[i]      = (__bf16)(x1 * c - x2 * s);
        dst[i + 32] = (__bf16)(x2 * c + x1 * s);
    } else {
        int hk = head - NUM_H;
        const __bf16* src = qkv + (size_t)t * QKV_N + Q_SZ + hk * HEAD_D;
        float x1 = (float)src[i], x2 = (float)src[i + 32];
        __bf16* dst = k_rope + (size_t)t * KV_SZ + hk * HEAD_D;
        dst[i]      = (__bf16)(x1 * c - x2 * s);
        dst[i + 32] = (__bf16)(x2 * c + x1 * s);
    }
}

// ---------------------------------------------------------------------------
// V transpose: v_t[kvh][d][t] = qkv[t][Q_SZ + KV_SZ + kvh*64 + d]
// Makes PV B-fragments contiguous-in-K (token) loads.
// ---------------------------------------------------------------------------
__global__ void v_transpose_kernel(const __bf16* __restrict__ qkv,
                                   __bf16* __restrict__ v_t) {
    int idx = blockIdx.x * blockDim.x + threadIdx.x;
    const int total = NUM_KV * HEAD_D * T_SEQ;
    if (idx >= total) return;
    int t   = idx % T_SEQ;
    int d   = (idx / T_SEQ) % HEAD_D;
    int kvh = idx / (T_SEQ * HEAD_D);
    v_t[idx] = qkv[(size_t)t * QKV_N + Q_SZ + KV_SZ + kvh * HEAD_D + d];
}

// ---------------------------------------------------------------------------
// Flash attention: one wave32 per (head, 16-query tile).
// Streams 32-key chunks: 4 WMMAs for S, online softmax via 16-lane shuffles,
// P redistributed C-layout -> A-layout through per-wave LDS (s_wait_dscnt),
// 4 WMMAs for P*V. Causal trip counts differ per wave -> no block barriers.
// ---------------------------------------------------------------------------
__global__ void __launch_bounds__(256)
attn_kernel(const __bf16* __restrict__ q_rope,
            const __bf16* __restrict__ k_rope,
            const __bf16* __restrict__ v_t,
            __bf16* __restrict__ out) {
    __shared__ __bf16 plds[8][16 * 32];   // 1 KB per wave, 8 KB per block

    const int lane = threadIdx.x & 31;
    const int lr = lane & 15;
    const int g  = lane >> 4;
    const int wib = threadIdx.x >> 5;
    const int wid = (blockIdx.x * blockDim.x + threadIdx.x) >> 5;
    const int QT = T_SEQ / 16;            // 128 query tiles
    const int h  = wid / QT;
    const int qt = wid % QT;
    if (h >= NUM_H) return;
    const int kvh = h >> 2;               // GQA: 4 Q heads per KV head
    const int qbase = qt * 16;

    // Q fragment (16 rows x 64 K) -- loop invariant.
    const __bf16* qb = q_rope + (size_t)(qbase + lr) * Q_SZ + h * HEAD_D + g * 8;
    v16bf aqA = load_ab_frag(qb);         // d 0..31
    v16bf aqB = load_ab_frag(qb + 32);    // d 32..63

    v8f o[4] = {};
    float m_run[8], l_run[8];
#pragma unroll
    for (int r = 0; r < 8; ++r) { m_run[r] = -__builtin_inff(); l_run[r] = 0.f; }

    __bf16* pl = &plds[wib][0];
    const int nchunks = (qbase + 16 + 31) / 32;

    for (int c = 0; c < nchunks; ++c) {
        const int kbase = c * 32;

        // K^T B-fragments: lane = key (N), runs over d (K).
        const __bf16* kp0 = k_rope + (size_t)(kbase + lr) * KV_SZ + kvh * HEAD_D + g * 8;
        const __bf16* kp1 = kp0 + (size_t)16 * KV_SZ;
        v16bf k00 = load_ab_frag(kp0);
        v16bf k01 = load_ab_frag(kp0 + 32);
        v16bf k10 = load_ab_frag(kp1);
        v16bf k11 = load_ab_frag(kp1 + 32);

        v8f S0 = {}, S1 = {};
        S0 = wmma_bf16(aqA, k00, S0);  S0 = wmma_bf16(aqB, k01, S0);
        S1 = wmma_bf16(aqA, k10, S1);  S1 = wmma_bf16(aqB, k11, S1);

        const bool need_mask = (kbase + 31) > qbase;

#pragma unroll
        for (int r = 0; r < 8; ++r) {
            float s0 = S0[r] * SM_SCALE;
            float s1 = S1[r] * SM_SCALE;
            if (need_mask) {
                int q = qbase + r + 8 * g;
                if (kbase + lr > q)      s0 = -3.0e38f;
                if (kbase + 16 + lr > q) s1 = -3.0e38f;
            }
            // Row stats: row m = r + 8g lives in VGPR r across a 16-lane half.
            float cm = rmax16(fmaxf(s0, s1));
            float mn = fmaxf(m_run[r], cm);
            float alpha = __expf(m_run[r] - mn);
            float p0 = __expf(s0 - mn);
            float p1 = __expf(s1 - mn);
            l_run[r] = l_run[r] * alpha + rsum16(p0 + p1);
            m_run[r] = mn;
            o[0][r] *= alpha; o[1][r] *= alpha; o[2][r] *= alpha; o[3][r] *= alpha;

            // Stage P (bf16) to LDS in [m][n] order for A-layout reload.
            int m = r + 8 * g;
            pl[m * 32 + lr]      = (__bf16)p0;
            pl[m * 32 + 16 + lr] = (__bf16)p1;
        }

        asm volatile("s_wait_dscnt 0" ::: "memory");   // CDNA5 split DS counter
        v16bf pf = load_ab_frag(pl + lr * 32 + g * 8); // P as 16x32 A fragment

        // V B-fragments from transposed V: contiguous in token (K dim).
        const __bf16* vb = v_t + (size_t)kvh * HEAD_D * T_SEQ + kbase + g * 8;
#pragma unroll
        for (int nt = 0; nt < 4; ++nt) {
            v16bf vf = load_ab_frag(vb + (size_t)(nt * 16 + lr) * T_SEQ);
            o[nt] = wmma_bf16(pf, vf, o[nt]);
        }
    }

    // Normalize and emit attn output [t][h*64 + d] as bf16.
#pragma unroll
    for (int nt = 0; nt < 4; ++nt)
#pragma unroll
        for (int r = 0; r < 8; ++r) {
            int t = qbase + r + 8 * g;
            int d = h * HEAD_D + nt * 16 + lr;
            out[(size_t)t * Q_SZ + d] = (__bf16)(o[nt][r] / l_run[r]);
        }
}

// ---------------------------------------------------------------------------
// Host-side orchestration
// ---------------------------------------------------------------------------
extern "C" void kernel_launch(void* const* d_in, const int* in_sizes, int n_in,
                              void* d_out, int out_size, void* d_ws, size_t ws_size,
                              hipStream_t stream) {
    const float* hidden    = (const float*)d_in[0];
    const int*   positions = (const int*)d_in[1];
    const float* w_qkv     = (const float*)d_in[2];
    const float* w_o       = (const float*)d_in[3];
    float* out = (float*)d_out;

    char* ws = (char*)d_ws;
    size_t off = 0;
    __bf16* hidden_b = (__bf16*)(ws + off); off += (size_t)T_SEQ * HIDDEN_W * 2;
    __bf16* wqkv_b   = (__bf16*)(ws + off); off += (size_t)QKV_N * HIDDEN_W * 2;
    __bf16* wo_b     = (__bf16*)(ws + off); off += (size_t)HIDDEN_W * Q_SZ * 2;
    __bf16* qkv_b    = (__bf16*)(ws + off); off += (size_t)T_SEQ * QKV_N * 2;
    __bf16* q_rope   = (__bf16*)(ws + off); off += (size_t)T_SEQ * Q_SZ * 2;
    __bf16* k_rope   = (__bf16*)(ws + off); off += (size_t)T_SEQ * KV_SZ * 2;
    __bf16* v_t      = (__bf16*)(ws + off); off += (size_t)NUM_KV * HEAD_D * T_SEQ * 2;
    __bf16* attn_b   = (__bf16*)(ws + off); off += (size_t)T_SEQ * Q_SZ * 2;

    // 1) fp32 -> bf16 casts
    {
        int n;
        n = T_SEQ * HIDDEN_W;
        cast_f32_to_bf16<<<(n + 255) / 256, 256, 0, stream>>>(hidden, hidden_b, n);
        n = QKV_N * HIDDEN_W;
        cast_f32_to_bf16<<<(n + 255) / 256, 256, 0, stream>>>(w_qkv, wqkv_b, n);
        n = HIDDEN_W * Q_SZ;
        cast_f32_to_bf16<<<(n + 255) / 256, 256, 0, stream>>>(w_o, wo_b, n);
    }

    // 2) QKV projection: [2048,3072] = hidden[2048,2048] @ w_qkv[3072,2048]^T
    {
        int waves = (T_SEQ / 32) * (QKV_N / 32);          // 64*96 = 6144
        gemm_bf16_nt<<<waves * 32 / 256, 256, 0, stream>>>(
            hidden_b, wqkv_b, nullptr, qkv_b, T_SEQ, QKV_N, HIDDEN_W);
    }

    // 3) RoPE on Q and K
    {
        int n = T_SEQ * (NUM_H + NUM_KV) * (HEAD_D / 2);
        rope_kernel<<<(n + 255) / 256, 256, 0, stream>>>(qkv_b, positions, q_rope, k_rope);
    }

    // 4) V transpose for PV fragment loads
    {
        int n = NUM_KV * HEAD_D * T_SEQ;
        v_transpose_kernel<<<(n + 255) / 256, 256, 0, stream>>>(qkv_b, v_t);
    }

    // 5) Flash attention (one wave per head x 16-query tile)
    {
        int waves = NUM_H * (T_SEQ / 16);                 // 32*128 = 4096
        attn_kernel<<<waves * 32 / 256, 256, 0, stream>>>(q_rope, k_rope, v_t, attn_b);
    }

    // 6) Output projection: out[2048,2048] = attn[2048,2048] @ w_o[2048,2048]^T
    {
        int waves = (T_SEQ / 32) * (HIDDEN_W / 32);       // 64*64 = 4096
        gemm_bf16_nt<<<waves * 32 / 256, 256, 0, stream>>>(
            attn_b, wo_b, out, nullptr, T_SEQ, HIDDEN_W, Q_SZ);
    }
}
